// GroundTruthPermutedGruCell_51135880626901
// MI455X (gfx1250) — matrix operand order repaired and checked by reading, not compiled, not run on hardware
//
#include <hip/hip_runtime.h>
#include <hip/hip_bf16.h>

// GRU cell with lower-triangular-masked weights.
// Phase 1: one-time f32->bf16 convert (x, hidden) + mask/transpose/convert weights.
// Phase 2: fused 6-GEMM bf16 WMMA (v_wmma_f32_16x16x32_bf16), double-buffered
//          async LDS fills (global_load_async_to_lds_b128 / s_wait_asynccnt),
//          gate epilogue. Fallback (small ws): fully fused f32-source kernel.

#define B_DIM 4096
#define H_DIM 2048
#define BM    128   // batch rows per block
#define BN    32    // hidden cols per block
#define KB    32    // K per stage (one WMMA K-step)
#define KP    40    // padded K stride in LDS (elements) -> conflict-free b128 frags

typedef __attribute__((ext_vector_type(16))) __bf16 v16bf;
typedef __attribute__((ext_vector_type(8)))  float  v8f;

union FragBF {
  v16bf v;
  uint4 q[2];
};

__device__ __forceinline__ unsigned f2bf_raw(float f) {
  unsigned u = __float_as_uint(f);
  u += 0x7fffu + ((u >> 16) & 1u);   // round-to-nearest-even
  return u >> 16;
}
__device__ __forceinline__ unsigned pack2bf(float a, float b) {
  return f2bf_raw(a) | (f2bf_raw(b) << 16);
}
__device__ __forceinline__ float fast_sigmoid(float x) {
  return 1.0f / (1.0f + __expf(-x));
}
__device__ __forceinline__ float fast_tanh(float x) {
  return 2.0f * fast_sigmoid(2.0f * x) - 1.0f;
}

// Generic pointers into the LDS aperture carry the LDS byte offset in addr[31:0].
__device__ __forceinline__ unsigned lds_off(const void* p) {
  return (unsigned)(uintptr_t)p;
}

// Async DMA: 16B global -> LDS per lane, tracked by ASYNCcnt.
__device__ __forceinline__ void async_copy_b128(unsigned lds_byte_off, const void* gptr) {
  asm volatile("global_load_async_to_lds_b128 %0, %1, off"
               :: "v"(lds_byte_off), "v"((unsigned long long)(uintptr_t)gptr)
               : "memory");
}
template <int N>
__device__ __forceinline__ void wait_asynccnt() {
  asm volatile("s_wait_asynccnt %0" :: "i"(N) : "memory");
}

// A fragment (16x32 bf16): lane L holds row M=L&15; K groups {8g..8g+7} and {16+8g..}
__device__ __forceinline__ v16bf load_fragA(const unsigned short* tile, int lrow, int aK) {
  FragBF f;
  const unsigned short* p = tile + lrow * KP;
  f.q[0] = *(const uint4*)(p + aK);
  f.q[1] = *(const uint4*)(p + aK + 16);
  return f.v;
}
// B fragment (32x16 bf16) from transposed [n][k] tile: lane L holds col N=L&15;
// K groups {8g..8g+7} and {16+8g..}, g = half-wave.
__device__ __forceinline__ v16bf load_fragB(const unsigned short* tile, int lrow, int bK) {
  FragBF f;
  const unsigned short* p = tile + lrow * KP;
  f.q[0] = *(const uint4*)(p + bK);
  f.q[1] = *(const uint4*)(p + bK + 16);
  return f.v;
}

// ---------------- Phase 1a: f32 -> bf16 bulk convert ----------------
__global__ __launch_bounds__(256) void cvt_bf16_kernel(const float* __restrict__ src,
                                                       unsigned short* __restrict__ dst) {
  const int i = (blockIdx.x * 256 + threadIdx.x) * 8;
  const float4 a = *(const float4*)(src + i);
  const float4 b = *(const float4*)(src + i + 4);
  uint4 o;
  o.x = pack2bf(a.x, a.y);
  o.y = pack2bf(a.z, a.w);
  o.z = pack2bf(b.x, b.y);
  o.w = pack2bf(b.z, b.w);
  *(uint4*)(dst + i) = o;
}

// ------- Phase 1b: mask (k>=n) + transpose + convert one weight matrix -------
// Output Wt[n][k] bf16; tiles entirely below the diagonal are never read -> skip.
__global__ __launch_bounds__(256) void wmask_transpose_kernel(const float* __restrict__ W,
                                                              unsigned short* __restrict__ Wt) {
  const int k0 = blockIdx.x * 32;
  const int n0 = blockIdx.y * 32;
  if (k0 + 32 <= n0) return;   // fully masked AND never read by phase 2
  __shared__ unsigned short t[32][36];
  const int tid = threadIdx.x;
  {
    const int kr = tid >> 3;
    const int nn = (tid & 7) * 4;
    const int gk = k0 + kr;
    const float4 v = *(const float4*)(W + (size_t)gk * H_DIM + n0 + nn);
    t[kr][nn + 0] = (gk >= n0 + nn + 0) ? (unsigned short)f2bf_raw(v.x) : (unsigned short)0;
    t[kr][nn + 1] = (gk >= n0 + nn + 1) ? (unsigned short)f2bf_raw(v.y) : (unsigned short)0;
    t[kr][nn + 2] = (gk >= n0 + nn + 2) ? (unsigned short)f2bf_raw(v.z) : (unsigned short)0;
    t[kr][nn + 3] = (gk >= n0 + nn + 3) ? (unsigned short)f2bf_raw(v.w) : (unsigned short)0;
  }
  __syncthreads();
  const int nr = tid >> 3;
  const int ks = (tid & 7) * 4;
  uint2 o;
  o.x = (unsigned)t[ks + 0][nr] | ((unsigned)t[ks + 1][nr] << 16);
  o.y = (unsigned)t[ks + 2][nr] | ((unsigned)t[ks + 3][nr] << 16);
  *(uint2*)(Wt + (size_t)(n0 + nr) * H_DIM + k0 + ks) = o;
}

// ---------------- Phase 2: bf16-source WMMA GEMM, double-buffered async fills ----------------
__global__ __launch_bounds__(256) void gru_wmma_bf16_kernel(
    const unsigned short* __restrict__ xb, const unsigned short* __restrict__ hb,
    const unsigned short* __restrict__ Wt,   // [6][H][H], [n][k] masked bf16
    const float* __restrict__ hidden,
    const float* __restrict__ b_hr, const float* __restrict__ b_hz,
    const float* __restrict__ b_hn, float* __restrict__ out) {
  __shared__ unsigned short Xs[2][BM * KP];
  __shared__ unsigned short Hs[2][BM * KP];
  __shared__ unsigned short WL[2][6 * BN * KP];

  const int tid   = threadIdx.x;
  const int lane  = tid & 31;
  const int lrow  = lane & 15;
  const int lhalf = lane >> 4;
  const int aK    = lhalf * 8;
  const int bK    = lhalf * 8;

  const int m0 = blockIdx.x * BM;
  const int n0 = blockIdx.y * BN;

  const int wave = tid >> 5;
  const int mw   = (wave & 3) * 32;
  const int nw   = (wave >> 2) * 16;

  // Per-thread fill coordinates (uniform every stage; only k0 changes).
  const int xrow = tid >> 2;               // 2 chunks: rows tid>>2 and (tid+256)>>2
  const int xkk  = (tid & 3) * 8;
  const unsigned xloff = (unsigned)(xrow * KP + xkk) * 2u;

  const unsigned xs_base[2] = {lds_off(Xs[0]), lds_off(Xs[1])};
  const unsigned hs_base[2] = {lds_off(Hs[0]), lds_off(Hs[1])};
  const unsigned wl_base[2] = {lds_off(WL[0]), lds_off(WL[1])};

  v8f acc[6][2] = {};

  // Issue the 7 async B128 copies for one K-stage.
  auto issue_fill = [&](int s, int k0) {
#pragma unroll
    for (int i = 0; i < 2; ++i) {
      const int c   = tid + i * 256;
      const int row = c >> 2;
      const int kk  = (c & 3) * 8;
      const unsigned loff = (unsigned)(row * KP + kk) * 2u;
      const size_t  goff  = (size_t)(m0 + row) * H_DIM + k0 + kk;
      async_copy_b128(xs_base[s] + loff, xb + goff);
      async_copy_b128(hs_base[s] + loff, hb + goff);
    }
#pragma unroll
    for (int i = 0; i < 3; ++i) {
      const int c    = tid + i * 256;
      const int w    = c >> 7;
      const int r    = c & 127;
      const int nrow = r >> 2;
      const int kk   = (r & 3) * 8;
      async_copy_b128(wl_base[s] + (unsigned)((w * BN + nrow) * KP + kk) * 2u,
                      Wt + ((size_t)w * H_DIM + (n0 + nrow)) * H_DIM + k0 + kk);
    }
  };

  // Triangular skip: lower[k,n] = (k >= n) kills every K-block below n0.
  int stage = 0;
  issue_fill(0, n0);
  for (int k0 = n0; k0 < H_DIM; k0 += KB, stage ^= 1) {
    // Prefetch next stage (clamped re-read on the last iteration keeps waits uniform).
    const int knext = (k0 + KB < H_DIM) ? (k0 + KB) : k0;
    issue_fill(stage ^ 1, knext);
    // Asyncs complete in order: <=7 outstanding => the 7 current-stage ops landed.
    wait_asynccnt<7>();
    __syncthreads();

    const unsigned short* Xc = Xs[stage];
    const unsigned short* Hc = Hs[stage];
    const unsigned short* Wc = WL[stage];

    v16bf aA[2][2];
    aA[0][0] = load_fragA(Xc + (mw + 0) * KP, lrow, aK);
    aA[0][1] = load_fragA(Xc + (mw + 16) * KP, lrow, aK);
    aA[1][0] = load_fragA(Hc + (mw + 0) * KP, lrow, aK);
    aA[1][1] = load_fragA(Hc + (mw + 16) * KP, lrow, aK);

    v16bf bfr[6];
#pragma unroll
    for (int g = 0; g < 6; ++g)
      bfr[g] = load_fragB(Wc + (g * BN + nw) * KP, lrow, bK);

#pragma unroll
    for (int g = 0; g < 6; ++g) {
      const int s = (g < 3) ? 0 : 1;
      acc[g][0] = __builtin_amdgcn_wmma_f32_16x16x32_bf16(
          false, aA[s][0], false, bfr[g], (short)0, acc[g][0], false, false);
      acc[g][1] = __builtin_amdgcn_wmma_f32_16x16x32_bf16(
          false, aA[s][1], false, bfr[g], (short)0, acc[g][1], false, false);
    }
    __syncthreads();   // all waves done reading 'stage' before it is refilled
  }

  const int gn = n0 + nw + lrow;
  const float bhr = b_hr[gn];
  const float bhz = b_hz[gn];
  const float bhn = b_hn[gn];

#pragma unroll
  for (int mi = 0; mi < 2; ++mi) {
#pragma unroll
    for (int r = 0; r < 8; ++r) {
      const int gm = m0 + mw + mi * 16 + r + 8 * lhalf;
      const float rt = fast_sigmoid(acc[0][mi][r] + acc[3][mi][r] + bhr);
      const float zt = fast_sigmoid(acc[1][mi][r] + acc[4][mi][r] + bhz);
      const float nt = fast_tanh(acc[2][mi][r] + rt * (acc[5][mi][r] + bhn));
      const float hp = hidden[gm * H_DIM + gn];
      out[gm * H_DIM + gn] = hp * zt + (1.0f - zt) * nt;
    }
  }
}

// ---------------- Fallback: fully fused, f32 sources ----------------
__global__ __launch_bounds__(256) void gru_wmma_fused_kernel(
    const float* __restrict__ x, const float* __restrict__ hidden,
    const float* __restrict__ W_ir, const float* __restrict__ W_iz,
    const float* __restrict__ W_in, const float* __restrict__ W_hr,
    const float* __restrict__ W_hz, const float* __restrict__ W_hn,
    const float* __restrict__ b_hr, const float* __restrict__ b_hz,
    const float* __restrict__ b_hn, float* __restrict__ out) {
  __shared__ unsigned short Xs[BM * KP];
  __shared__ unsigned short Hs[BM * KP];
  __shared__ unsigned short WB[6][BN * KP];

  const int tid   = threadIdx.x;
  const int lane  = tid & 31;
  const int lrow  = lane & 15;
  const int lhalf = lane >> 4;
  const int aK    = lhalf * 8;
  const int bK    = lhalf * 8;

  const int m0 = blockIdx.x * BM;
  const int n0 = blockIdx.y * BN;

  const int wave = tid >> 5;
  const int mw   = (wave & 3) * 32;
  const int nw   = (wave >> 2) * 16;

  const float* Wsrc[6] = {W_ir, W_iz, W_in, W_hr, W_hz, W_hn};

  v8f acc[6][2] = {};

  for (int k0 = n0; k0 < H_DIM; k0 += KB) {
    __syncthreads();
#pragma unroll
    for (int i = 0; i < 4; ++i) {
      const int e   = (tid + i * 256) * 4;
      const int row = e >> 5;
      const int kk  = e & 31;
      const float4 xv = *(const float4*)(x + (m0 + row) * H_DIM + k0 + kk);
      const float4 hv = *(const float4*)(hidden + (m0 + row) * H_DIM + k0 + kk);
      uint2 px; px.x = pack2bf(xv.x, xv.y); px.y = pack2bf(xv.z, xv.w);
      uint2 ph; ph.x = pack2bf(hv.x, hv.y); ph.y = pack2bf(hv.z, hv.w);
      *(uint2*)&Xs[row * KP + kk] = px;
      *(uint2*)&Hs[row * KP + kk] = ph;
    }
#pragma unroll
    for (int w = 0; w < 6; ++w) {
      const int kr = (tid >> 5) << 2;
      const int nn = tid & 31;
      const int gk = k0 + kr;
      const int gn = n0 + nn;
      const float* p = Wsrc[w] + gk * H_DIM + gn;
      const float f0 = (gk + 0 >= gn) ? p[0 * H_DIM] : 0.0f;
      const float f1 = (gk + 1 >= gn) ? p[1 * H_DIM] : 0.0f;
      const float f2 = (gk + 2 >= gn) ? p[2 * H_DIM] : 0.0f;
      const float f3 = (gk + 3 >= gn) ? p[3 * H_DIM] : 0.0f;
      uint2 q; q.x = pack2bf(f0, f1); q.y = pack2bf(f2, f3);
      *(uint2*)&WB[w][nn * KP + kr] = q;
    }
    __syncthreads();

    v16bf ax0 = load_fragA(Xs + (mw + 0) * KP, lrow, aK);
    v16bf ax1 = load_fragA(Xs + (mw + 16) * KP, lrow, aK);
    v16bf ah0 = load_fragA(Hs + (mw + 0) * KP, lrow, aK);
    v16bf ah1 = load_fragA(Hs + (mw + 16) * KP, lrow, aK);

#pragma unroll
    for (int g = 0; g < 6; ++g) {
      const v16bf bb = load_fragB(&WB[g][nw * KP], lrow, bK);
      const v16bf a0 = (g < 3) ? ax0 : ah0;
      const v16bf a1 = (g < 3) ? ax1 : ah1;
      acc[g][0] = __builtin_amdgcn_wmma_f32_16x16x32_bf16(
          false, a0, false, bb, (short)0, acc[g][0], false, false);
      acc[g][1] = __builtin_amdgcn_wmma_f32_16x16x32_bf16(
          false, a1, false, bb, (short)0, acc[g][1], false, false);
    }
  }

  const int gn = n0 + nw + lrow;
  const float bhr = b_hr[gn];
  const float bhz = b_hz[gn];
  const float bhn = b_hn[gn];

#pragma unroll
  for (int mi = 0; mi < 2; ++mi) {
#pragma unroll
    for (int r = 0; r < 8; ++r) {
      const int gm = m0 + mw + mi * 16 + r + 8 * lhalf;
      const float rt = fast_sigmoid(acc[0][mi][r] + acc[3][mi][r] + bhr);
      const float zt = fast_sigmoid(acc[1][mi][r] + acc[4][mi][r] + bhz);
      const float nt = fast_tanh(acc[2][mi][r] + rt * (acc[5][mi][r] + bhn));
      const float hp = hidden[gm * H_DIM + gn];
      out[gm * H_DIM + gn] = hp * zt + (1.0f - zt) * nt;
    }
  }
}

extern "C" void kernel_launch(void* const* d_in, const int* in_sizes, int n_in,
                              void* d_out, int out_size, void* d_ws, size_t ws_size,
                              hipStream_t stream) {
  (void)in_sizes; (void)n_in; (void)out_size;
  const float* x      = (const float*)d_in[0];
  const float* hidden = (const float*)d_in[1];
  // d_in[2] = lower: tril(ones) -> mask applied analytically as (k >= n)
  const float* W_ir = (const float*)d_in[3];
  const float* W_hr = (const float*)d_in[4];
  const float* W_iz = (const float*)d_in[5];
  const float* W_hz = (const float*)d_in[6];
  const float* W_in = (const float*)d_in[7];
  const float* W_hn = (const float*)d_in[8];
  const float* b_hr = (const float*)d_in[9];
  const float* b_hz = (const float*)d_in[10];
  const float* b_hn = (const float*)d_in[11];
  float* out = (float*)d_out;

  const size_t XB_BYTES = (size_t)B_DIM * H_DIM * 2;        // 16 MB
  const size_t WT_BYTES = (size_t)6 * H_DIM * H_DIM * 2;    // 48 MB
  const size_t WS_NEEDED = 2 * XB_BYTES + WT_BYTES;         // 80 MB

  dim3 grid(B_DIM / BM, H_DIM / BN);
  dim3 block(256);

  if (ws_size >= WS_NEEDED) {
    unsigned short* xb = (unsigned short*)d_ws;
    unsigned short* hb = xb + (size_t)B_DIM * H_DIM;
    unsigned short* Wt = hb + (size_t)B_DIM * H_DIM;

    const int n_xh = B_DIM * H_DIM;
    cvt_bf16_kernel<<<n_xh / (256 * 8), 256, 0, stream>>>(x, xb);
    cvt_bf16_kernel<<<n_xh / (256 * 8), 256, 0, stream>>>(hidden, hb);

    const float* Wsrc[6] = {W_ir, W_iz, W_in, W_hr, W_hz, W_hn};
    dim3 tgrid(H_DIM / 32, H_DIM / 32);
    for (int w = 0; w < 6; ++w) {
      wmask_transpose_kernel<<<tgrid, 256, 0, stream>>>(
          Wsrc[w], Wt + (size_t)w * H_DIM * H_DIM);
    }

    gru_wmma_bf16_kernel<<<grid, block, 0, stream>>>(
        xb, hb, Wt, hidden, b_hr, b_hz, b_hn, out);
  } else {
    gru_wmma_fused_kernel<<<grid, block, 0, stream>>>(
        x, hidden, W_ir, W_iz, W_in, W_hr, W_hz, W_hn, b_hr, b_hz, b_hn, out);
  }
}
